// AttentiveLISTA_17471926960564
// MI455X (gfx1250) — matrix operand order repaired and checked by reading, not compiled, not run on hardware
//
#include <hip/hip_runtime.h>
#include <hip/hip_bf16.h>

typedef __bf16 bf16_t;
typedef __bf16 v16bf __attribute__((ext_vector_type(16)));
typedef __bf16 v8bf  __attribute__((ext_vector_type(8)));
typedef float  v8f   __attribute__((ext_vector_type(8)));
typedef unsigned short v8us __attribute__((ext_vector_type(8)));
typedef unsigned int v4u __attribute__((ext_vector_type(4)));
typedef int v4i __attribute__((ext_vector_type(4)));
typedef int v8i __attribute__((ext_vector_type(8)));

#define P_TOT   32768      // B*H*W = 8*64*64
#define HW_TOT  4096
#define W_DIM   64
#define H_DIM   64
#define B_DIM   8
#define LISTA_ROWS 32      // rows of z per workgroup

#if __has_builtin(__builtin_amdgcn_tensor_load_to_lds) && \
    __has_builtin(__builtin_amdgcn_s_wait_tensorcnt)
#define USE_TDM 1
#else
#define USE_TDM 0
#endif

// ---------------- WMMA helpers ----------------

__device__ inline v8f wmma_bf16(v16bf a, v16bf b, v8f c) {
  return __builtin_amdgcn_wmma_f32_16x16x32_bf16(
      /*neg_a=*/false, a, /*neg_b=*/false, b,
      /*c_mod=*/(short)0, c, /*reuse_a=*/false, /*reuse_b=*/false);
}

__device__ inline v8bf relu8(v8bf x) {
  v8us u = __builtin_bit_cast(v8us, x);
#pragma unroll
  for (int i = 0; i < 8; i++) u[i] = (unsigned short)((u[i] & 0x8000u) ? 0u : u[i]);
  return __builtin_bit_cast(v8bf, u);
}

__device__ inline v16bf cat16(v8bf lo, v8bf hi) {
  return __builtin_shufflevector(lo, hi, 0,1,2,3,4,5,6,7,8,9,10,11,12,13,14,15);
}

// A fragment (16x32 bf16) for one lane from a row-major bf16 row pointer.
// lane: m = lane&15, half = lane>>4. K set = {8*half+0..7} U {16+8*half+0..7}.
template <bool RELU>
__device__ inline v16bf load_a_row(const bf16_t* row, int colbase, int half, bool valid) {
  if (valid) {
    v8bf lo = *(const v8bf*)(row + colbase + 8 * half);
    v8bf hi = *(const v8bf*)(row + colbase + 16 + 8 * half);
    if (RELU) { lo = relu8(lo); hi = relu8(hi); }
    return cat16(lo, hi);
  }
  v16bf z;
#pragma unroll
  for (int i = 0; i < 16; i++) z[i] = (bf16_t)0.0f;
  return z;
}

// B fragment (32x16 bf16) from B stored transposed: blay[n][k], row stride Ks.
// lane: n = lane&15, half = lane>>4; reads k at kb + 8*half + 0..7 and kb + 16 + 8*half + 0..7.
__device__ inline v16bf load_b_t(const bf16_t* blay, int n, int Ks, int kb, int half) {
  const bf16_t* rp = blay + (size_t)n * Ks + kb + 8 * half;
  v8bf lo = *(const v8bf*)(rp);
  v8bf hi = *(const v8bf*)(rp + 16);
  return cat16(lo, hi);
}

// ---------------- prep kernels ----------------

__global__ void k_xln(const float* __restrict__ x, bf16_t* __restrict__ xln) {
  size_t t = (size_t)blockIdx.x * 256 + threadIdx.x;  // P*64 threads
  int c = (int)(t & 63);
  size_t p = t >> 6;
  int b = (int)(p >> 12);
  int hw = (int)(p & 4095);
  xln[t] = (bf16_t)x[((size_t)(b * 64 + c)) * HW_TOT + hw];
}

__global__ void k_wT(const float* __restrict__ w, bf16_t* __restrict__ wT,
                     int Cout, int Cin, int ksq) {
  int t = blockIdx.x * 256 + threadIdx.x;
  int total = Cout * Cin * ksq;
  if (t >= total) return;
  int co = t / (Cin * ksq);
  int rem = t % (Cin * ksq);
  int ci = rem / ksq;
  int k = rem % ksq;
  wT[((size_t)k * Cout + co) * Cin + ci] = (bf16_t)w[t];
}

__global__ void k_dictprep(const float* __restrict__ Dict, bf16_t* __restrict__ dict_bf,
                           bf16_t* __restrict__ dictT_bf, float* __restrict__ out_dict) {
  int t = blockIdx.x * 256 + threadIdx.x;  // 16384 threads
  int c = t >> 8, a = t & 255;
  float v = Dict[t];
  dict_bf[t] = (bf16_t)v;           // [c][a]  (B-layout for recon: N=c, K=a)
  dictT_bf[a * 64 + c] = (bf16_t)v; // [a][c]  (B-layout for y: N=a, K=c)
  out_dict[t] = v;
}

__global__ void k_S(const float* __restrict__ Dict, const float* __restrict__ Lp,
                    bf16_t* __restrict__ Sbf) {
  int t = blockIdx.x * 256 + threadIdx.x;  // 65536 threads
  int i = t >> 8, j = t & 255;
  float s = 0.f;
#pragma unroll 4
  for (int c = 0; c < 64; c++) s += Dict[c * 256 + i] * Dict[c * 256 + j];
  float v = ((i == j) ? 1.0f : 0.0f) - s / (*Lp);
  Sbf[t] = (bf16_t)v;  // S symmetric: row-major == [n][k]
}

// ---------------- generic implicit-GEMM conv (WMMA) ----------------
// in: bf16 [P, Cin] channels-last.  wT: bf16 [(ky*ks+kx)][co][ci].
// OUT_MODE: 0 = bf16 [P,Cout], 1 = f32 [P,Cout], 2 = f32 NCHW-transposed.
template <bool RELU_IN, bool HAS_BIAS, bool HAS_RES, int OUT_MODE>
__global__ void conv_wmma(const bf16_t* __restrict__ in, const bf16_t* __restrict__ wT,
                          const float* __restrict__ bias, const bf16_t* __restrict__ res,
                          void* __restrict__ out, int Cin, int Cout, int ks, int pad) {
  int wid = (blockIdx.x * blockDim.x + threadIdx.x) >> 5;
  int lane = threadIdx.x & 31;
  int ntiles = Cout >> 4;
  int msegs = P_TOT >> 4;
  if (wid >= msegs * ntiles) return;
  int mseg = wid / ntiles;
  int nt = wid % ntiles;
  int p0 = mseg << 4;
  int b = p0 >> 12;
  int hw = p0 & 4095;
  int h = hw >> 6;
  int w0 = hw & 63;
  int n = lane & 15;   // also the A row index m within the tile
  int half = lane >> 4;
  int n0 = nt << 4;

  v8f acc;
#pragma unroll
  for (int r = 0; r < 8; r++) acc[r] = 0.f;

  for (int ky = 0; ky < ks; ky++) {
    int hh = h + ky - pad;
    bool hok = (hh >= 0) && (hh < H_DIM);
    for (int kx = 0; kx < ks; kx++) {
      int ww = w0 + n + kx - pad;  // per-lane pixel column (m = n = lane&15)
      bool ok = hok && (ww >= 0) && (ww < W_DIM);
      const bf16_t* rowp = in + ((size_t)((b * H_DIM + hh) * W_DIM + ww)) * Cin;
      const bf16_t* bl = wT + (size_t)(ky * ks + kx) * Cout * Cin;
      for (int kc = 0; kc < Cin; kc += 32) {
        v16bf a = load_a_row<RELU_IN>(rowp, kc, half, ok);
        v16bf bb = load_b_t(bl, n0 + n, Cin, kc, half);
        acc = wmma_bf16(a, bb, acc);
      }
    }
  }

  float bv = HAS_BIAS ? bias[n0 + n] : 0.f;
#pragma unroll
  for (int r = 0; r < 8; r++) {
    int m = r + 8 * half;
    int p = p0 + m;
    size_t idx = (size_t)p * Cout + n0 + n;
    float v = acc[r] + bv;
    if (HAS_RES) v += (float)res[idx];
    if (OUT_MODE == 0) {
      ((bf16_t*)out)[idx] = (bf16_t)v;
    } else if (OUT_MODE == 1) {
      ((float*)out)[idx] = v;
    } else {
      int bb2 = p >> 12, hw2 = p & 4095;
      ((float*)out)[(size_t)(bb2 * Cout + n0 + n) * HW_TOT + hw2] = v;
    }
  }
}

// ---------------- attention kernels (tiny, VALU) ----------------

__global__ void k_pool1(const bf16_t* __restrict__ l3, float* __restrict__ pavg,
                        float* __restrict__ pmax) {
  int bi = blockIdx.x;            // 8*32 blocks
  int b = bi >> 5, chunk = bi & 31;
  int a = threadIdx.x;            // 256 channels
  float s = 0.f, m = -1e30f;
  size_t base = ((size_t)b * HW_TOT + chunk * 128) * 256;
  for (int j = 0; j < 128; j++) {
    float v = (float)l3[base + (size_t)j * 256 + a];
    s += v;
    m = fmaxf(m, v);
  }
  pavg[bi * 256 + a] = s;
  pmax[bi * 256 + a] = m;
}

__global__ void k_ca(const float* __restrict__ pavg, const float* __restrict__ pmax,
                     const float* __restrict__ w1, const float* __restrict__ w2,
                     float* __restrict__ sca) {
  __shared__ float sA[256], sM[256], hA[16], hM[16];
  int b = blockIdx.x;
  int a = threadIdx.x;
  float s = 0.f, m = -1e30f;
  for (int c = 0; c < 32; c++) {
    s += pavg[(b * 32 + c) * 256 + a];
    m = fmaxf(m, pmax[(b * 32 + c) * 256 + a]);
  }
  sA[a] = s * (1.0f / 4096.0f);
  sM[a] = m;
  __syncthreads();
  if (a < 16) {
    float da = 0.f, dm = 0.f;
    for (int c = 0; c < 256; c++) {
      float wv = w1[a * 256 + c];
      da += sA[c] * wv;
      dm += sM[c] * wv;
    }
    hA[a] = fmaxf(da, 0.f);
    hM[a] = fmaxf(dm, 0.f);
  }
  __syncthreads();
  float o = 0.f;
#pragma unroll
  for (int j = 0; j < 16; j++) o += (hA[j] + hM[j]) * w2[a * 16 + j];
  sca[b * 256 + a] = 1.f / (1.f + expf(-o));
}

__global__ void k_sp1(const bf16_t* __restrict__ l3, const float* __restrict__ sca,
                      float* __restrict__ mn, float* __restrict__ mxm) {
  int p = blockIdx.x * 256 + threadIdx.x;  // P threads
  int b = p >> 12;
  float s = 0.f, m = -1e30f;
  const bf16_t* row = l3 + (size_t)p * 256;
  const float* sc = sca + b * 256;
  for (int a = 0; a < 256; a++) {
    float v = (float)row[a] * sc[a];
    s += v;
    m = fmaxf(m, v);
  }
  mn[p] = s * (1.0f / 256.0f);
  mxm[p] = m;
}

__global__ void k_sp2(const float* __restrict__ mn, const float* __restrict__ mxm,
                      const float* __restrict__ sw, float* __restrict__ sa) {
  int p = blockIdx.x * 256 + threadIdx.x;
  int b = p >> 12, hw = p & 4095, h = hw >> 6, w = hw & 63;
  float acc = 0.f;
#pragma unroll
  for (int ky = 0; ky < 3; ky++) {
    int hh = h + ky - 1;
    if (hh < 0 || hh >= H_DIM) continue;
#pragma unroll
    for (int kx = 0; kx < 3; kx++) {
      int ww = w + kx - 1;
      if (ww < 0 || ww >= W_DIM) continue;
      int q = (b << 12) + (hh << 6) + ww;
      acc += mn[q] * sw[(0 * 3 + ky) * 3 + kx] + mxm[q] * sw[(1 * 3 + ky) * 3 + kx];
    }
  }
  sa[p] = 1.f / (1.f + expf(-acc));
}

__global__ void k_lam(const bf16_t* __restrict__ l3, const float* __restrict__ sca,
                      const float* __restrict__ sa, const float* __restrict__ Lp,
                      float* __restrict__ lam) {
  size_t t = (size_t)blockIdx.x * 256 + threadIdx.x;  // P*256 threads
  int a = (int)(t & 255);
  size_t p = t >> 8;
  int b = (int)(p >> 12);
  lam[t] = (float)l3[t] * sca[b * 256 + a] * sa[p] / (*Lp);
}

// ---------------- fused 16-iteration LISTA (WMMA + TDM) ----------------
// Workgroup (8 waves) owns 32 rows of z. S (bf16, 128 KB) is DMA'd into LDS
// by the Tensor Data Mover (overlapped with z0 init), z tile (16 KB) in LDS.
// Waves = 2 M-tiles x 4 N-quads; 4 f32 accumulator tiles per wave (no spills).
__global__ __launch_bounds__(256)
void k_lista(const float* __restrict__ y, const float* __restrict__ lam,
             const bf16_t* __restrict__ S, const int* __restrict__ niter_p,
             const float* __restrict__ Lp, float* __restrict__ out_z,
             bf16_t* __restrict__ zbf) {
  __shared__ bf16_t Ssh[256 * 256];          // 128 KB
  __shared__ bf16_t zsh[LISTA_ROWS * 256];   // 16 KB
  int tid = threadIdx.x;
  int lane = tid & 31;
  int wv = tid >> 5;            // 0..7
  int mt = wv >> 2;             // M tile within WG (0..1)
  int nq = wv & 3;              // N quad (4 N-tiles of 16 each)
  int m0 = blockIdx.x * LISTA_ROWS + mt * 16;
  int n = lane & 15;
  int half = lane >> 4;
  int niter = *niter_p;
  float invL = 1.0f / (*Lp);

#if USE_TDM
  // TDM: DMA S (128 KB = 32768 dwords) global -> LDS; one descriptor, wave 0.
  // D# group0: count=1, lds_addr, global_addr, type=2.
  // D# group1: data_size=4B, tensor_dim0=tile_dim0=32768 dwords, dims1=1,
  //            tensor_dim0_stride=32768, no pad/iterate/multicast.
  if (wv == 0) {
    unsigned lds_off =
        (unsigned)(size_t)(__attribute__((address_space(3))) bf16_t*)Ssh;
    unsigned long long ga = (unsigned long long)(const void*)S;
    v4u g0 = { 1u, lds_off, (unsigned)ga, (unsigned)(ga >> 32) | 0x80000000u };
    v8i g1 = { 0x20000, (int)0x80000000, 0x00010000, (int)0x80000000,
               1, 32768, 0, 0 };
    v4i gz = { 0, 0, 0, 0 };
#if defined(__clang_major__) && (__clang_major__ >= 23)
    v8i gz8 = { 0, 0, 0, 0, 0, 0, 0, 0 };
    __builtin_amdgcn_tensor_load_to_lds(g0, g1, gz, gz, gz8, 0);
#else
    __builtin_amdgcn_tensor_load_to_lds(g0, g1, gz, gz, 0);
#endif
  }
#else
  // Fallback: cooperative load S -> LDS (each thread: 32 x b128)
  {
    const v8bf* src = (const v8bf*)S;
    v8bf* dst = (v8bf*)Ssh;
#pragma unroll 4
    for (int i = tid; i < (256 * 256) / 8; i += 256) dst[i] = src[i];
  }
#endif

  // z0 = soft(y) for this block's 32 rows (overlaps with the TDM transfer)
  for (int i = tid; i < LISTA_ROWS * 256; i += 256) {
    size_t idx = (size_t)blockIdx.x * LISTA_ROWS * 256 + i;
    float v = y[idx];
    float av = fabsf(v) - lam[idx];
    zsh[i] = (bf16_t)((av > 0.f) ? copysignf(av, v) : 0.f);
  }
#if USE_TDM
  if (wv == 0) __builtin_amdgcn_s_wait_tensorcnt(0);
#endif
  __syncthreads();

  const bf16_t* arow = zsh + (size_t)(mt * 16 + n) * 256;  // A row (m = n = lane&15)
  v8f acc[4];

  for (int t = 0; t < niter; t++) {
#pragma unroll
    for (int q = 0; q < 4; q++)
#pragma unroll
      for (int r = 0; r < 8; r++) acc[q][r] = 0.f;

    for (int kc = 0; kc < 256; kc += 32) {
      v16bf a = cat16(*(const v8bf*)(arow + kc + 8 * half),
                      *(const v8bf*)(arow + kc + 16 + 8 * half));
#pragma unroll
      for (int q = 0; q < 4; q++) {
        int nn = (nq * 4 + q) * 16 + n;
        const bf16_t* rp = Ssh + (size_t)nn * 256 + kc + 8 * half;
        v16bf bb = cat16(*(const v8bf*)rp, *(const v8bf*)(rp + 16));
        acc[q] = wmma_bf16(a, bb, acc[q]);
      }
    }
    __syncthreads();  // all waves done reading zsh (full K) before writeback

    bool last = (t == niter - 1);
#pragma unroll
    for (int q = 0; q < 4; q++) {
      int nn0 = (nq * 4 + q) * 16;
#pragma unroll
      for (int r = 0; r < 8; r++) {
        int m = r + 8 * half;
        size_t idx = (size_t)(m0 + m) * 256 + nn0 + n;
        float v = acc[q][r] + y[idx] * invL;
        float av = fabsf(v) - lam[idx];
        v = (av > 0.f) ? copysignf(av, v) : 0.f;
        if (!last) {
          zsh[(mt * 16 + m) * 256 + nn0 + n] = (bf16_t)v;
        } else {
          zbf[idx] = (bf16_t)v;
          int p = m0 + m;
          int bb2 = p >> 12, hw2 = p & 4095;
          out_z[(size_t)(bb2 * 256 + nn0 + n) * HW_TOT + hw2] = v;
        }
      }
    }
    __syncthreads();  // writeback visible before next iteration's reads
  }
}

// ---------------- host ----------------

extern "C" void kernel_launch(void* const* d_in, const int* in_sizes, int n_in,
                              void* d_out, int out_size, void* d_ws, size_t ws_size,
                              hipStream_t stream) {
  const float* x      = (const float*)d_in[0];
  const float* conv_w = (const float*)d_in[1];
  const float* conv_b = (const float*)d_in[2];
  const float* r1w1   = (const float*)d_in[3];
  const float* r1w2   = (const float*)d_in[4];
  const float* r2w1   = (const float*)d_in[5];
  const float* r2w2   = (const float*)d_in[6];
  const float* caw1   = (const float*)d_in[7];
  const float* caw2   = (const float*)d_in[8];
  const float* saw    = (const float*)d_in[9];
  const float* Dict   = (const float*)d_in[10];
  const float* Lp     = (const float*)d_in[11];
  const int*   itp    = (const int*)d_in[12];

  float* out_z    = (float*)d_out;                              // [8,256,64,64]
  float* out_xr   = out_z + (size_t)B_DIM * 256 * HW_TOT;       // [8,64,64,64]
  float* out_dict = out_xr + (size_t)B_DIM * 64 * HW_TOT;       // [64,256]

  char* ws = (char*)d_ws;
  size_t off = 0;
  auto alloc = [&](size_t bytes) -> char* {
    char* p = ws + off;
    off += (bytes + 255) & ~((size_t)255);
    return p;
  };

  bf16_t* xln     = (bf16_t*)alloc((size_t)P_TOT * 64 * 2);
  bf16_t* w1T     = (bf16_t*)alloc((size_t)9 * 256 * 64 * 2);
  bf16_t* r1w1T   = (bf16_t*)alloc((size_t)9 * 128 * 256 * 2);
  bf16_t* r1w2T   = (bf16_t*)alloc((size_t)256 * 128 * 2);
  bf16_t* r2w1T   = (bf16_t*)alloc((size_t)9 * 128 * 256 * 2);
  bf16_t* r2w2T   = (bf16_t*)alloc((size_t)256 * 128 * 2);
  bf16_t* dict_bf = (bf16_t*)alloc((size_t)64 * 256 * 2);
  bf16_t* dictT_bf= (bf16_t*)alloc((size_t)256 * 64 * 2);
  bf16_t* Sbf     = (bf16_t*)alloc((size_t)256 * 256 * 2);
  bf16_t* l1      = (bf16_t*)alloc((size_t)P_TOT * 256 * 2);
  bf16_t* tbuf    = (bf16_t*)alloc((size_t)P_TOT * 128 * 2);
  bf16_t* l2      = (bf16_t*)alloc((size_t)P_TOT * 256 * 2);
  bf16_t* l3      = (bf16_t*)alloc((size_t)P_TOT * 256 * 2);
  float*  pavg    = (float*)alloc((size_t)8 * 32 * 256 * 4);
  float*  pmax    = (float*)alloc((size_t)8 * 32 * 256 * 4);
  float*  sca     = (float*)alloc((size_t)8 * 256 * 4);
  float*  mn      = (float*)alloc((size_t)P_TOT * 4);
  float*  mxm     = (float*)alloc((size_t)P_TOT * 4);
  float*  sa      = (float*)alloc((size_t)P_TOT * 4);
  float*  ybuf    = (float*)alloc((size_t)P_TOT * 256 * 4);
  float*  lam     = (float*)alloc((size_t)P_TOT * 256 * 4);
  bf16_t* zbf     = (bf16_t*)alloc((size_t)P_TOT * 256 * 2);
  (void)ws_size; (void)n_in; (void)in_sizes; (void)out_size;

  // --- prep ---
  k_xln<<<(P_TOT * 64) / 256, 256, 0, stream>>>(x, xln);
  k_wT<<<(256 * 64 * 9) / 256, 256, 0, stream>>>(conv_w, w1T, 256, 64, 9);
  k_wT<<<(128 * 256 * 9) / 256, 256, 0, stream>>>(r1w1, r1w1T, 128, 256, 9);
  k_wT<<<(256 * 128) / 256, 256, 0, stream>>>(r1w2, r1w2T, 256, 128, 1);
  k_wT<<<(128 * 256 * 9) / 256, 256, 0, stream>>>(r2w1, r2w1T, 128, 256, 9);
  k_wT<<<(256 * 128) / 256, 256, 0, stream>>>(r2w2, r2w2T, 256, 128, 1);
  k_dictprep<<<16384 / 256, 256, 0, stream>>>(Dict, dict_bf, dictT_bf, out_dict);
  k_S<<<65536 / 256, 256, 0, stream>>>(Dict, Lp, Sbf);

  const int msegs = P_TOT / 16;  // 2048

  // --- conv front-end (implicit GEMM, WMMA) ---
  conv_wmma<false, true, false, 0><<<(msegs * 16) / 4, 128, 0, stream>>>(
      xln, w1T, conv_b, nullptr, l1, 64, 256, 3, 1);
  conv_wmma<true, false, false, 0><<<(msegs * 8) / 4, 128, 0, stream>>>(
      l1, r1w1T, nullptr, nullptr, tbuf, 256, 128, 3, 1);
  conv_wmma<true, false, true, 0><<<(msegs * 16) / 4, 128, 0, stream>>>(
      tbuf, r1w2T, nullptr, l1, l2, 128, 256, 1, 0);
  conv_wmma<true, false, false, 0><<<(msegs * 8) / 4, 128, 0, stream>>>(
      l2, r2w1T, nullptr, nullptr, tbuf, 256, 128, 3, 1);
  conv_wmma<true, false, true, 0><<<(msegs * 16) / 4, 128, 0, stream>>>(
      tbuf, r2w2T, nullptr, l2, l3, 128, 256, 1, 0);

  // --- CBAM ---
  k_pool1<<<8 * 32, 256, 0, stream>>>(l3, pavg, pmax);
  k_ca<<<8, 256, 0, stream>>>(pavg, pmax, caw1, caw2, sca);
  k_sp1<<<P_TOT / 256, 256, 0, stream>>>(l3, sca, mn, mxm);
  k_sp2<<<P_TOT / 256, 256, 0, stream>>>(mn, mxm, saw, sa);
  k_lam<<<(P_TOT * 256) / 256, 256, 0, stream>>>(l3, sca, sa, Lp, lam);

  // --- y = x @ Dict (WMMA GEMM, K=64, N=256) ---
  conv_wmma<false, false, false, 1><<<(msegs * 16) / 4, 128, 0, stream>>>(
      xln, dictT_bf, nullptr, nullptr, ybuf, 64, 256, 1, 0);

  // --- fused LISTA iterations (S DMA'd to LDS via TDM) ---
  k_lista<<<P_TOT / LISTA_ROWS, 256, 0, stream>>>(ybuf, lam, Sbf, itp, Lp, out_z, zbf);

  // --- x_recon = z @ Dict^T (WMMA GEMM, K=256, N=64), transposed NCHW store ---
  conv_wmma<false, false, false, 2><<<(msegs * 4) / 4, 128, 0, stream>>>(
      zbf, dict_bf, nullptr, nullptr, out_xr, 256, 64, 1, 0);
}
